// GraphAttentionLayer_23252952940600
// MI455X (gfx1250) — matrix operand order repaired and checked by reading, not compiled, not run on hardware
//
#include <hip/hip_runtime.h>
#include <hip/hip_bf16.h>

// ---------------------------------------------------------------------------
// GAT layer for MI455X (gfx1250, wave32, WMMA).
// B=8, N=2048, F_in=F_out=128.
//   Wh = h @ W^T            (bf16 WMMA, f32 accum)
//   f = Wh a1, g = Wh a2
//   e = leakyrelu(f_i+g_j) masked by adj ; softmax over j (online/flash style)
//   out = elu(attn @ Wh)    (bf16 WMMA, f32 accum, fused with softmax)
//
// Roofline: adj (134 MB, NT-streamed once) -> ~5.8 us @ 23.3 TB/s HBM.
// Wh tiles staged in LDS per block (8 waves share) to cut L2->WGP fragment
// traffic 8x (512 MB -> 64 MB). Wh/f/g stay L2-resident (NT policy on adj).
// ---------------------------------------------------------------------------

typedef __attribute__((ext_vector_type(16))) __bf16 v16bf;
typedef __attribute__((ext_vector_type(8)))  float  v8f;
typedef __attribute__((ext_vector_type(4)))  int    v4i_;

union BF16Frag {
  v16bf v;
  uint4 q[2];
  unsigned short s[16];
};

__device__ __forceinline__ unsigned short f2bf(float x) {
  unsigned int u = __float_as_uint(x);
  unsigned int r = u + 0x7FFFu + ((u >> 16) & 1u);   // round-to-nearest-even
  return (unsigned short)(r >> 16);
}

#define B_   8
#define N_   2048
#define F_   128
#define NEG_INF_ (-9.0e15f)
#define LDS_STRIDE 40   // 32 bf16 + 8 pad -> 80 B row stride = 20 banks

// ---------------------------------------------------------------------------
// Kernel 0: fp32 -> bf16 conversion of h and W into workspace
// ---------------------------------------------------------------------------
__global__ void gat_convert_kernel(const float* __restrict__ h,
                                   const float* __restrict__ W,
                                   unsigned short* __restrict__ hbf,
                                   unsigned short* __restrict__ Wbf) {
  const int NH = B_ * N_ * F_;           // 2,097,152
  const int NW = F_ * F_;                // 16,384
  int idx = blockIdx.x * blockDim.x + threadIdx.x;
  if (idx < NH) {
    hbf[idx] = f2bf(h[idx]);
  } else if (idx < NH + NW) {
    int k = idx - NH;
    Wbf[k] = f2bf(W[k]);
  }
}

// ---------------------------------------------------------------------------
// Kernel 1: Wh = h @ W^T  via v_wmma_f32_16x16x32_bf16.
// One wave owns a 16-row n-tile. Also computes f = Wh a1, g = Wh a2 via
// in-register cross-lane reductions, and stores Wh transposed ([b][o][n],
// bf16) so kernel 2's B-fragments are contiguous 32-byte runs.
// ---------------------------------------------------------------------------
__global__ __launch_bounds__(256)
void gat_wh_kernel(const unsigned short* __restrict__ hbf,
                   const unsigned short* __restrict__ Wbf,
                   const float* __restrict__ a1,
                   const float* __restrict__ a2,
                   unsigned short* __restrict__ WhT,
                   float* __restrict__ fvec,
                   float* __restrict__ gvec) {
  const int lane  = threadIdx.x & 31;
  const int wave  = threadIdx.x >> 5;
  const int tile  = blockIdx.x * 8 + wave;          // B*N/16 = 1024 tiles
  const int b     = tile >> 7;                      // tile / (N/16)
  const int nbase = (tile & 127) << 4;
  const int half  = lane >> 4;                      // K-half select
  const int lrow  = lane & 15;

  // A fragments: 16 h-rows x K=128 in 4 chunks of 32.
  // 16-bit A 16x32 layout: half 0 -> K = {0..7, 16..23}; half 1 -> +8.
  BF16Frag afrag[4];
  const unsigned short* hrow = hbf + (size_t)(b * N_ + nbase + lrow) * F_;
#pragma unroll
  for (int kc = 0; kc < 4; ++kc) {
    const int kb = kc * 32 + half * 8;
    afrag[kc].q[0] = *(const uint4*)(hrow + kb);        // elems 0..7
    afrag[kc].q[1] = *(const uint4*)(hrow + kb + 16);   // elems 8..15
  }

  float fp[8], gp[8];
#pragma unroll
  for (int r = 0; r < 8; ++r) { fp[r] = 0.f; gp[r] = 0.f; }

#pragma unroll
  for (int ot = 0; ot < 8; ++ot) {                  // 8 o-tiles of 16 outputs
    const int obase = ot * 16;
    // B fragment: B[k][n] = W[obase+lrow][k]; K = {0..7,16..23} + half*8 (+kc*32).
    const unsigned short* wrow = Wbf + (size_t)(obase + lrow) * F_;

    v8f acc;
#pragma unroll
    for (int r = 0; r < 8; ++r) acc[r] = 0.f;

#pragma unroll
    for (int kc = 0; kc < 4; ++kc) {
      const int kb = kc * 32 + half * 8;
      BF16Frag bfrag;
      bfrag.q[0] = *(const uint4*)(wrow + kb);
      bfrag.q[1] = *(const uint4*)(wrow + kb + 16);
      acc = __builtin_amdgcn_wmma_f32_16x16x32_bf16(
          false, afrag[kc].v, false, bfrag.v, (short)0, acc, false, false);
    }

    // C layout: lanes 0-15 hold rows r (N=lane); lanes 16-31 rows 8+r.
    const float a1v = a1[obase + lrow];
    const float a2v = a2[obase + lrow];
#pragma unroll
    for (int r = 0; r < 8; ++r) {
      const float v = acc[r];
      fp[r] += v * a1v;
      gp[r] += v * a2v;
      const int nrow = nbase + r + half * 8;
      WhT[(size_t)(b * F_ + obase + lrow) * N_ + nrow] = f2bf(v);
    }
  }

  // Reduce f/g across the 16 lanes (columns) of each half.
#pragma unroll
  for (int r = 0; r < 8; ++r) {
    float fv = fp[r], gv = gp[r];
#pragma unroll
    for (int msk = 1; msk < 16; msk <<= 1) {
      fv += __shfl_xor(fv, msk, 32);
      gv += __shfl_xor(gv, msk, 32);
    }
    if (lrow == 0) {
      const int nrow = nbase + r + half * 8;
      fvec[b * N_ + nrow] = fv;
      gvec[b * N_ + nrow] = gv;
    }
  }
}

// ---------------------------------------------------------------------------
// Kernel 2: fused masked-softmax + (attn @ Wh) + ELU, flash-attention style.
// 8 waves/block own 8 consecutive i-tiles of the same batch and walk the
// j-chunks in lockstep, sharing each 8KB Wh[j-chunk] slab via double-buffered
// LDS (one barrier per chunk). adj streamed once with NT cache policy.
// ---------------------------------------------------------------------------
__global__ __launch_bounds__(256)
void gat_attn_kernel(const int* __restrict__ adj,
                     const unsigned short* __restrict__ WhT,
                     const float* __restrict__ fvec,
                     const float* __restrict__ gvec,
                     float* __restrict__ out) {
  __shared__ unsigned short ldsW[2][F_ * LDS_STRIDE];   // 2 x 10 KB

  const int tid   = threadIdx.x;
  const int lane  = tid & 31;
  const int wave  = tid >> 5;
  const int tile  = blockIdx.x * 8 + wave;          // 1024 i-tiles
  const int b     = tile >> 7;                      // uniform across block
  const int ibase = (tile & 127) << 4;
  const int half  = lane >> 4;
  const int lrow  = lane & 15;

  // LDS fill: 256 threads move 8 KB (WhT[b][0..127][jbase..jbase+31]).
  // Thread t handles o-row t/2, 16-element segment t&1.
  const int fo  = tid >> 1;
  const int fsg = (tid & 1) * 16;
  const unsigned short* fsrc = WhT + (size_t)(b * F_ + fo) * N_ + fsg;
  unsigned short* fdst = &ldsW[0][0] + fo * LDS_STRIDE + fsg;
  const int bufElems = F_ * LDS_STRIDE;

#define FILL(bufidx, jcn)                                              \
  do {                                                                 \
    const unsigned short* s_ = fsrc + (jcn) * 32;                      \
    uint4 d0_ = *(const uint4*)(s_);                                   \
    uint4 d1_ = *(const uint4*)(s_ + 8);                               \
    unsigned short* d_ = fdst + (bufidx) * bufElems;                   \
    *(uint4*)(d_) = d0_;                                               \
    *(uint4*)(d_ + 8) = d1_;                                           \
  } while (0)

  const float fi = fvec[b * N_ + ibase + lrow];     // this lane's softmax row
  const float* grow = gvec + (size_t)b * N_;
  const int* adjrow = adj + (size_t)(b * N_ + ibase + lrow) * N_;

  v8f acc[8];
#pragma unroll
  for (int ot = 0; ot < 8; ++ot)
#pragma unroll
    for (int r = 0; r < 8; ++r) acc[ot][r] = 0.f;

  float m = -INFINITY;   // running row max (duplicated in lane pair {i,i+16})
  float l = 0.f;         // running row sum

  FILL(0, 0);

  for (int jc = 0; jc < N_ / 32; ++jc) {
    __syncthreads();                       // fill of buf (jc&1) visible
    if (jc + 1 < N_ / 32) FILL((jc + 1) & 1, jc + 1);

    const int jbase = jc * 32;
    const int jA = jbase + half * 8;        // K elems 0..7
    const int jB = jbase + 16 + half * 8;   // K elems 8..15

    // Prefetch next chunk's adj (speculative, streamed once).
    __builtin_prefetch(adjrow + jA + 32, 0, 0);

    float ge[16]; int ad[16];
    *(float4*)(ge + 0)  = *(const float4*)(grow + jA);
    *(float4*)(ge + 4)  = *(const float4*)(grow + jA + 4);
    *(float4*)(ge + 8)  = *(const float4*)(grow + jB);
    *(float4*)(ge + 12) = *(const float4*)(grow + jB + 4);
    // adj: read-once 134 MB -> non-temporal so Wh/f/g stay L2-resident.
    *(v4i_*)(ad + 0)  = __builtin_nontemporal_load((const v4i_*)(adjrow + jA));
    *(v4i_*)(ad + 4)  = __builtin_nontemporal_load((const v4i_*)(adjrow + jA + 4));
    *(v4i_*)(ad + 8)  = __builtin_nontemporal_load((const v4i_*)(adjrow + jB));
    *(v4i_*)(ad + 12) = __builtin_nontemporal_load((const v4i_*)(adjrow + jB + 4));

    float e[16];
    float cmax = -INFINITY;
#pragma unroll
    for (int t = 0; t < 16; ++t) {
      float x = fi + ge[t];
      x = (x > 0.f) ? x : 0.2f * x;               // leaky_relu(0.2)
      x = (ad[t] > 0) ? x : NEG_INF_;             // adjacency mask
      e[t] = x;
      cmax = fmaxf(cmax, x);
    }
    cmax = fmaxf(cmax, __shfl_xor(cmax, 16, 32)); // join K-halves of the row
    const float mnew  = fmaxf(m, cmax);
    const float alpha = __expf(m - mnew);         // first iter: exp(-inf)=0
    m = mnew;

    BF16Frag pfrag;                               // probs as A fragment
    float s = 0.f;
#pragma unroll
    for (int t = 0; t < 16; ++t) {
      const float p = __expf(e[t] - mnew);
      s += p;
      pfrag.s[t] = f2bf(p);
    }
    s += __shfl_xor(s, 16, 32);
    l = l * alpha + s;

    // Rescale accumulators: C-row M = r + half*8; alpha for row M lives in
    // lane M -> full-crossbar shuffle (ds_bpermute).
    float aR[8];
#pragma unroll
    for (int r = 0; r < 8; ++r) aR[r] = __shfl(alpha, r + half * 8, 32);
#pragma unroll
    for (int ot = 0; ot < 8; ++ot)
#pragma unroll
      for (int r = 0; r < 8; ++r) acc[ot][r] *= aR[r];

    // B fragments from LDS (conflict-free: 80B row stride = 20 banks).
    // Lane needs Wh[jbase + half*8 + {0..7,16..23}][ot*16+lrow].
    const unsigned short* lbase =
        &ldsW[0][0] + (jc & 1) * bufElems + lrow * LDS_STRIDE + half * 8;
    BF16Frag bf[8];
#pragma unroll
    for (int ot = 0; ot < 8; ++ot) {
      const unsigned short* lp = lbase + ot * 16 * LDS_STRIDE;
      bf[ot].q[0] = *(const uint4*)(lp);
      bf[ot].q[1] = *(const uint4*)(lp + 16);
    }
#pragma unroll
    for (int ot = 0; ot < 8; ++ot) {
      acc[ot] = __builtin_amdgcn_wmma_f32_16x16x32_bf16(
          false, pfrag.v, false, bf[ot].v, (short)0, acc[ot], false, false);
    }
  }

  // Finalize: divide by row sum, ELU, store fp32 (NT, write-once).
  const float linv = 1.0f / l;
  float lR[8];
#pragma unroll
  for (int r = 0; r < 8; ++r) lR[r] = __shfl(linv, r + half * 8, 32);

#pragma unroll
  for (int ot = 0; ot < 8; ++ot) {
#pragma unroll
    for (int r = 0; r < 8; ++r) {
      float x = acc[ot][r] * lR[r];
      x = (x > 0.f) ? x : expm1f(x);              // elu(alpha=1)
      const int irow = ibase + r + half * 8;
      __builtin_nontemporal_store(
          x, out + (size_t)(b * N_ + irow) * F_ + ot * 16 + lrow);
    }
  }
#undef FILL
}

// ---------------------------------------------------------------------------
// Host launcher. Inputs: h[8,2048,128] f32, adj[8,2048,2048] i32,
// W[128,128] f32, a1[128] f32, a2[128] f32. Output: [8,2048,128] f32.
// ---------------------------------------------------------------------------
extern "C" void kernel_launch(void* const* d_in, const int* in_sizes, int n_in,
                              void* d_out, int out_size, void* d_ws, size_t ws_size,
                              hipStream_t stream) {
  const float* h   = (const float*)d_in[0];
  const int*   adj = (const int*)d_in[1];
  const float* W   = (const float*)d_in[2];
  const float* a1  = (const float*)d_in[3];
  const float* a2  = (const float*)d_in[4];
  float* out = (float*)d_out;

  // Workspace layout (bytes):
  //   hbf  : 0          .. 4,194,304   (2,097,152 bf16)
  //   Wbf  : 4,194,304  .. 4,227,072   (16,384 bf16)
  //   WhT  : 4,227,072  .. 8,421,376   (2,097,152 bf16, [b][o][n])
  //   fvec : 8,421,376  .. 8,486,912   (16,384 f32)
  //   gvec : 8,486,912  .. 8,552,448   (16,384 f32)
  char* ws = (char*)d_ws;
  unsigned short* hbf  = (unsigned short*)(ws + 0);
  unsigned short* Wbf  = (unsigned short*)(ws + 4194304);
  unsigned short* WhT  = (unsigned short*)(ws + 4227072);
  float*          fvec = (float*)(ws + 8421376);
  float*          gvec = (float*)(ws + 8486912);

  const int nconv = B_ * N_ * F_ + F_ * F_;        // 2,113,536
  gat_convert_kernel<<<(nconv + 255) / 256, 256, 0, stream>>>(h, W, hbf, Wbf);

  // B*N/16 = 1024 waves, 8 waves (256 threads) per block -> 128 blocks.
  gat_wh_kernel<<<128, 256, 0, stream>>>(hbf, Wbf, a1, a2, WhT, fvec, gvec);
  gat_attn_kernel<<<128, 256, 0, stream>>>(adj, WhT, fvec, gvec, out);
}
